// TransformerEncoderLayer_76845554860242
// MI455X (gfx1250) — matrix operand order repaired and checked by reading, not compiled, not run on hardware
//
#include <hip/hip_runtime.h>
#include <hip/hip_bf16.h>

typedef __hip_bfloat16 bf16_t;
typedef __attribute__((ext_vector_type(16))) __bf16 v16bf;
typedef __attribute__((ext_vector_type(8)))  float  v8f;

union Frag16 { v16bf v; uint4 u[2]; };

constexpr int Bc    = 2;
constexpr int Sc    = 2048;
constexpr int Dc    = 1024;
constexpr int Hc    = 16;
constexpr int HDc   = 64;
constexpr int FFc   = 4096;
constexpr int MAXSc = 5000;

__device__ __forceinline__ v8f wmma_bf16(v16bf a, v16bf b, v8f c) {
  // D = A(16x32 bf16) * B(32x16 bf16) + C(16x16 f32)
  return __builtin_amdgcn_wmma_f32_16x16x32_bf16(false, a, false, b, (short)0, c,
                                                 false, false);
}

__device__ __forceinline__ float red_max16(float x) {
  x = fmaxf(x, __shfl_xor(x, 1, 32));
  x = fmaxf(x, __shfl_xor(x, 2, 32));
  x = fmaxf(x, __shfl_xor(x, 4, 32));
  x = fmaxf(x, __shfl_xor(x, 8, 32));
  return x;
}
__device__ __forceinline__ float red_sum16(float x) {
  x += __shfl_xor(x, 1, 32);
  x += __shfl_xor(x, 2, 32);
  x += __shfl_xor(x, 4, 32);
  x += __shfl_xor(x, 8, 32);
  return x;
}

// ---------------------------------------------------------------------------
// Weight convert + transpose: Wt[n*K + k] = bf16(W[k*N + n])
// ---------------------------------------------------------------------------
__global__ void wcvt_kernel(const float* __restrict__ W, bf16_t* __restrict__ Wt,
                            int K, int N) {
  size_t idx = (size_t)blockIdx.x * 256 + threadIdx.x;
  int kk = (int)(idx / N);
  int n  = (int)(idx % N);
  Wt[(size_t)n * K + kk] = __float2bfloat16(W[idx]);
}

// ---------------------------------------------------------------------------
// LayerNorm (one row of 1024 per block), output bf16
// ---------------------------------------------------------------------------
__global__ __launch_bounds__(256) void layernorm_kernel(
    const float* __restrict__ x, const float* __restrict__ g,
    const float* __restrict__ be, bf16_t* __restrict__ out) {
  const int row  = blockIdx.x;
  const int t    = threadIdx.x;
  const int lane = t & 31, wid = t >> 5;
  const float4* xr = (const float4*)(x + (size_t)row * Dc);
  float4 v = xr[t];

  __shared__ float red[8];
  __shared__ float stats[2];

  float s = v.x + v.y + v.z + v.w;
#pragma unroll
  for (int o = 16; o > 0; o >>= 1) s += __shfl_xor(s, o, 32);
  if (lane == 0) red[wid] = s;
  __syncthreads();
  if (t == 0) {
    float tot = 0.f;
#pragma unroll
    for (int i = 0; i < 8; ++i) tot += red[i];
    stats[0] = tot * (1.0f / Dc);
  }
  __syncthreads();
  const float mu = stats[0];
  const float dx = v.x - mu, dy = v.y - mu, dz = v.z - mu, dw = v.w - mu;
  float ss = dx * dx + dy * dy + dz * dz + dw * dw;
#pragma unroll
  for (int o = 16; o > 0; o >>= 1) ss += __shfl_xor(ss, o, 32);
  if (lane == 0) red[wid] = ss;
  __syncthreads();
  if (t == 0) {
    float tot = 0.f;
#pragma unroll
    for (int i = 0; i < 8; ++i) tot += red[i];
    stats[1] = rsqrtf(tot * (1.0f / Dc) + 1e-5f);
  }
  __syncthreads();
  const float rstd = stats[1];
  float4 gg = ((const float4*)g)[t];
  float4 bb = ((const float4*)be)[t];
  bf16_t* orow = out + (size_t)row * Dc + t * 4;
  orow[0] = __float2bfloat16(dx * rstd * gg.x + bb.x);
  orow[1] = __float2bfloat16(dy * rstd * gg.y + bb.y);
  orow[2] = __float2bfloat16(dz * rstd * gg.z + bb.z);
  orow[3] = __float2bfloat16(dw * rstd * gg.w + bb.w);
}

// ---------------------------------------------------------------------------
// bf16 WMMA GEMM: C[M,N] = A[M,K] @ BT[N,K]^T + bias, fused epilogue.
//   EPI 0: bf16 store, val = scale*(acc+bias)            (Q/K proj; Q folds 1/8)
//   EPI 1: bf16 store, val = gelu(acc+bias)              (FFN up)
//   EPI 2: f32 store,  val = resid + acc + bias          (attn-out / FFN down)
//   EPI 3: bf16 store transposed to Vt[(b,h,d),s]        (V projection)
// Block: 256 threads = 8 waves; block tile 128x128; wave tile 32x64; K-step 32.
// ---------------------------------------------------------------------------
template <int EPI>
__global__ __launch_bounds__(256) void gemm_wmma_kernel(
    const bf16_t* __restrict__ A, const bf16_t* __restrict__ BT,
    const float* __restrict__ bias, void* __restrict__ Cout,
    const float* __restrict__ resid, float scale, int M, int N, int K) {
  __shared__ bf16_t As[128][40];  // 80B row stride -> conflict-free b128 reads
  __shared__ bf16_t Bs[128][40];

  const int t    = threadIdx.x;
  const int w    = t >> 5;
  const int lane = t & 31;
  const int hL   = lane >> 4;   // which 16-lane half
  const int lc   = lane & 15;
  const int wm   = w & 3;       // 4 waves along M
  const int wn   = w >> 2;      // 2 waves along N
  const int m0   = blockIdx.y * 128;
  const int n0   = blockIdx.x * 128;
  const int lr   = t >> 1;      // 0..127: tile row loaded by this thread
  const int seg  = t & 1;       // 16-element segment within K-step

  const bf16_t* arow = A  + (size_t)(m0 + lr) * K + seg * 16;
  const bf16_t* brow = BT + (size_t)(n0 + lr) * K + seg * 16;

  const v8f zero8 = {0.f, 0.f, 0.f, 0.f, 0.f, 0.f, 0.f, 0.f};
  v8f acc[2][4];
#pragma unroll
  for (int mt = 0; mt < 2; ++mt)
#pragma unroll
    for (int nt = 0; nt < 4; ++nt) acc[mt][nt] = zero8;

  for (int k0 = 0; k0 < K; k0 += 32) {
    __syncthreads();
    {
      const uint4* ga = (const uint4*)(arow + k0);
      *(uint4*)&As[lr][seg * 16]     = ga[0];
      *(uint4*)&As[lr][seg * 16 + 8] = ga[1];
      const uint4* gb = (const uint4*)(brow + k0);
      *(uint4*)&Bs[lr][seg * 16]     = gb[0];
      *(uint4*)&Bs[lr][seg * 16 + 8] = gb[1];
      // software prefetch of next K-step (lowers to global_prefetch_b8)
      if (k0 + 32 < K) {
        __builtin_prefetch(arow + k0 + 32, 0, 1);
        __builtin_prefetch(brow + k0 + 32, 0, 1);
      }
    }
    __syncthreads();

    Frag16 a[2], b[4];
#pragma unroll
    for (int mt = 0; mt < 2; ++mt) {
      const int r = wm * 32 + mt * 16 + lc;
      a[mt].u[0] = *(const uint4*)&As[r][8 * hL];        // K = 8*hL..+7
      a[mt].u[1] = *(const uint4*)&As[r][16 + 8 * hL];   // K = 16+8*hL..+7
    }
#pragma unroll
    for (int nt = 0; nt < 4; ++nt) {
      const int c = wn * 64 + nt * 16 + lc;
      b[nt].u[0] = *(const uint4*)&Bs[c][8 * hL];        // K = 8*hL..+7
      b[nt].u[1] = *(const uint4*)&Bs[c][16 + 8 * hL];   // K = 16+8*hL..+7
    }
#pragma unroll
    for (int mt = 0; mt < 2; ++mt)
#pragma unroll
      for (int nt = 0; nt < 4; ++nt)
        acc[mt][nt] = wmma_bf16(a[mt].v, b[nt].v, acc[mt][nt]);
  }

  // Epilogue: lane lc holds column, vgpr v holds row v + 8*hL of each tile.
#pragma unroll
  for (int mt = 0; mt < 2; ++mt) {
#pragma unroll
    for (int nt = 0; nt < 4; ++nt) {
      const int gcol = n0 + wn * 64 + nt * 16 + lc;
      const float bv = bias[gcol];
#pragma unroll
      for (int v = 0; v < 8; ++v) {
        const int grow = m0 + wm * 32 + mt * 16 + v + 8 * hL;
        const float val = acc[mt][nt][v] + bv;
        const size_t idx = (size_t)grow * N + gcol;
        if constexpr (EPI == 0) {
          ((bf16_t*)Cout)[idx] = __float2bfloat16(val * scale);
        } else if constexpr (EPI == 1) {
          const float gel = 0.5f * val * (1.0f + erff(val * 0.70710678118f));
          ((bf16_t*)Cout)[idx] = __float2bfloat16(gel);
        } else if constexpr (EPI == 2) {
          ((float*)Cout)[idx] = resid[idx] + val;
        } else {  // EPI == 3: V -> Vt[((b*H + h)*64 + d)*S + s]
          const int bb2 = grow >> 11, s2 = grow & (Sc - 1);
          const int hh = gcol >> 6, dd = gcol & (HDc - 1);
          ((bf16_t*)Cout)[(size_t)((bb2 * Hc + hh) * HDc + dd) * Sc + s2] =
              __float2bfloat16(val);
        }
      }
    }
  }
}

// ---------------------------------------------------------------------------
// Flash attention, one (b,h) per blockIdx.y, 64 q-rows per block (16 per wave).
// Q pre-scaled by 1/sqrt(64). K row-major (B,S,D); V pre-transposed (B,H,64,S).
// ---------------------------------------------------------------------------
__global__ __launch_bounds__(128) void attn_kernel(
    const bf16_t* __restrict__ Q, const bf16_t* __restrict__ K,
    const bf16_t* __restrict__ Vt, const int* __restrict__ mask,
    const float* __restrict__ bt, bf16_t* __restrict__ ctx) {
  const int w    = threadIdx.x >> 5;
  const int lane = threadIdx.x & 31;
  const int hL   = lane >> 4;
  const int lc   = lane & 15;
  const int bh   = blockIdx.y;
  const int b    = bh >> 4;
  const int h    = bh & 15;
  const int qb   = blockIdx.x * 64 + w * 16;

  __shared__ bf16_t Plds[4][16][40];  // per-wave P relayout buffer

  // Q A-fragments (row = lc, head dims split 2x32)
  Frag16 qa[2];
  {
    const bf16_t* qrow = Q + ((size_t)(b * Sc + qb + lc)) * Dc + h * HDc;
#pragma unroll
    for (int c = 0; c < 2; ++c) {
      qa[c].u[0] = *(const uint4*)(qrow + c * 32 + 8 * hL);
      qa[c].u[1] = *(const uint4*)(qrow + c * 32 + 16 + 8 * hL);
    }
  }

  const v8f zero8 = {0.f, 0.f, 0.f, 0.f, 0.f, 0.f, 0.f, 0.f};
  v8f o[4];
#pragma unroll
  for (int nt = 0; nt < 4; ++nt) o[nt] = zero8;
  float mrow[8], lrow[8];
#pragma unroll
  for (int v = 0; v < 8; ++v) { mrow[v] = -1e30f; lrow[v] = 0.f; }

  const int* mbase = mask + (size_t)b * Sc * Sc;
  const bf16_t* kbase  = K  + ((size_t)(b * Sc)) * Dc + h * HDc;
  const bf16_t* vtbase = Vt + ((size_t)((b * Hc + h) * HDc)) * Sc;

  for (int jb = 0; jb < Sc; jb += 32) {
    // prefetch next key/value tiles (lowers to global_prefetch_b8)
    if (jb + 32 < Sc) {
      __builtin_prefetch(kbase + (size_t)(jb + 32 + lane) * Dc, 0, 1);
      __builtin_prefetch(vtbase + (size_t)(lane << 1) * Sc + jb + 32, 0, 1);
    }

    // scores: S[16q x 32k] as two 16x16 fragments, sum over head dim 64
    v8f sc[2];
#pragma unroll
    for (int fr = 0; fr < 2; ++fr) {
      const int kb = jb + fr * 16;
      const bf16_t* krow = kbase + (size_t)(kb + lc) * Dc;
      sc[fr] = zero8;
#pragma unroll
      for (int c = 0; c < 2; ++c) {
        Frag16 kf;
        kf.u[0] = *(const uint4*)(krow + c * 32 + 8 * hL);
        kf.u[1] = *(const uint4*)(krow + c * 32 + 16 + 8 * hL);
        sc[fr]  = wmma_bf16(qa[c].v, kf.v, sc[fr]);
      }
    }

    // bias + mask + online softmax; write P (bf16) to LDS for relayout
#pragma unroll
    for (int v = 0; v < 8; ++v) {
      const int q = qb + v + 8 * hL;
      int k0 = jb + lc;
      int r0 = min(max(k0 - q + (MAXSc - 1), 0), 2 * MAXSc - 2);
      float s0 = sc[0][v] + bt[r0 * Hc + h];
      if (mbase[(size_t)q * Sc + k0] == 0) s0 = -1e9f;
      int k1 = jb + 16 + lc;
      int r1 = min(max(k1 - q + (MAXSc - 1), 0), 2 * MAXSc - 2);
      float s1 = sc[1][v] + bt[r1 * Hc + h];
      if (mbase[(size_t)q * Sc + k1] == 0) s1 = -1e9f;

      const float tmax = red_max16(fmaxf(s0, s1));
      const float mn   = fmaxf(mrow[v], tmax);
      const float alpha = __expf(mrow[v] - mn);
      mrow[v] = mn;
      const float p0 = __expf(s0 - mn);
      const float p1 = __expf(s1 - mn);
      lrow[v] = lrow[v] * alpha + red_sum16(p0 + p1);
#pragma unroll
      for (int nt = 0; nt < 4; ++nt) o[nt][v] = o[nt][v] * alpha;
      Plds[w][v + 8 * hL][lc]      = __float2bfloat16(p0);
      Plds[w][v + 8 * hL][lc + 16] = __float2bfloat16(p1);
    }

    // same-wave LDS RAW: wait the CDNA5 split DS counter, then read A-fragment
    asm volatile("s_wait_dscnt 0" ::: "memory");
    Frag16 pa;
    pa.u[0] = *(const uint4*)&Plds[w][lc][8 * hL];
    pa.u[1] = *(const uint4*)&Plds[w][lc][16 + 8 * hL];

    // O += P(16x32) @ V(32x64): B-fragments contiguous from transposed Vt
#pragma unroll
    for (int nt = 0; nt < 4; ++nt) {
      const bf16_t* vtr = vtbase + (size_t)(nt * 16 + lc) * Sc + jb + 8 * hL;
      Frag16 vf;
      vf.u[0] = *(const uint4*)vtr;
      vf.u[1] = *(const uint4*)(vtr + 16);
      o[nt]   = wmma_bf16(pa.v, vf.v, o[nt]);
    }
  }

  // normalize and store ctx (B,S,D) bf16
#pragma unroll
  for (int v = 0; v < 8; ++v) {
    const float inv = 1.0f / lrow[v];
    const int q = qb + v + 8 * hL;
    bf16_t* crow = ctx + ((size_t)(b * Sc + q)) * Dc + h * HDc;
#pragma unroll
    for (int nt = 0; nt < 4; ++nt)
      crow[nt * 16 + lc] = __float2bfloat16(o[nt][v] * inv);
  }
}

// ---------------------------------------------------------------------------
extern "C" void kernel_launch(void* const* d_in, const int* in_sizes, int n_in,
                              void* d_out, int out_size, void* d_ws, size_t ws_size,
                              hipStream_t stream) {
  (void)in_sizes; (void)n_in; (void)out_size; (void)ws_size;

  const float* x    = (const float*)d_in[0];
  const int*   mask = (const int*)d_in[1];
  const float* wq = (const float*)d_in[2];  const float* bq = (const float*)d_in[3];
  const float* wk = (const float*)d_in[4];  const float* bk = (const float*)d_in[5];
  const float* wv = (const float*)d_in[6];  const float* bv = (const float*)d_in[7];
  const float* wo = (const float*)d_in[8];  const float* bo = (const float*)d_in[9];
  const float* w1 = (const float*)d_in[10]; const float* b1 = (const float*)d_in[11];
  const float* w2 = (const float*)d_in[12]; const float* b2 = (const float*)d_in[13];
  const float* g1 = (const float*)d_in[14]; const float* be1 = (const float*)d_in[15];
  const float* g2 = (const float*)d_in[16]; const float* be2 = (const float*)d_in[17];
  const float* bt = (const float*)d_in[18];
  float* out = (float*)d_out;

  char* ws = (char*)d_ws;
  size_t off = 0;
  auto take = [&](size_t bytes) -> char* {
    char* p = ws + off;
    off += (bytes + 255) & ~(size_t)255;
    return p;
  };
  const int M = Bc * Sc;  // 4096
  bf16_t* wqT = (bf16_t*)take((size_t)Dc * Dc * 2);
  bf16_t* wkT = (bf16_t*)take((size_t)Dc * Dc * 2);
  bf16_t* wvT = (bf16_t*)take((size_t)Dc * Dc * 2);
  bf16_t* woT = (bf16_t*)take((size_t)Dc * Dc * 2);
  bf16_t* w1T = (bf16_t*)take((size_t)Dc * FFc * 2);
  bf16_t* w2T = (bf16_t*)take((size_t)FFc * Dc * 2);
  bf16_t* xn  = (bf16_t*)take((size_t)M * Dc * 2);
  bf16_t* Qb  = (bf16_t*)take((size_t)M * Dc * 2);
  bf16_t* Kb  = (bf16_t*)take((size_t)M * Dc * 2);
  bf16_t* VtB = (bf16_t*)take((size_t)M * Dc * 2);
  bf16_t* ctx = (bf16_t*)take((size_t)M * Dc * 2);
  float*  x1  = (float*)take((size_t)M * Dc * 4);
  bf16_t* xn2 = (bf16_t*)take((size_t)M * Dc * 2);
  bf16_t* hb  = (bf16_t*)take((size_t)M * FFc * 2);

  // weight convert + transpose (bf16, N-major -> K-contiguous B operands)
  wcvt_kernel<<<(Dc * Dc) / 256, 256, 0, stream>>>(wq, wqT, Dc, Dc);
  wcvt_kernel<<<(Dc * Dc) / 256, 256, 0, stream>>>(wk, wkT, Dc, Dc);
  wcvt_kernel<<<(Dc * Dc) / 256, 256, 0, stream>>>(wv, wvT, Dc, Dc);
  wcvt_kernel<<<(Dc * Dc) / 256, 256, 0, stream>>>(wo, woT, Dc, Dc);
  wcvt_kernel<<<(Dc * FFc) / 256, 256, 0, stream>>>(w1, w1T, Dc, FFc);
  wcvt_kernel<<<(FFc * Dc) / 256, 256, 0, stream>>>(w2, w2T, FFc, Dc);

  // LN1 -> xn (bf16)
  layernorm_kernel<<<M, 256, 0, stream>>>(x, g1, be1, xn);

  // Q/K/V projections (Q folds 1/sqrt(64); V stored transposed)
  gemm_wmma_kernel<0><<<dim3(Dc / 128, M / 128), 256, 0, stream>>>(
      xn, wqT, bq, Qb, nullptr, 0.125f, M, Dc, Dc);
  gemm_wmma_kernel<0><<<dim3(Dc / 128, M / 128), 256, 0, stream>>>(
      xn, wkT, bk, Kb, nullptr, 1.0f, M, Dc, Dc);
  gemm_wmma_kernel<3><<<dim3(Dc / 128, M / 128), 256, 0, stream>>>(
      xn, wvT, bv, VtB, nullptr, 1.0f, M, Dc, Dc);

  // flash attention with rel-pos bias + mask
  attn_kernel<<<dim3(Sc / 64, Bc * Hc), 128, 0, stream>>>(Qb, Kb, VtB, mask, bt, ctx);

  // output projection + residual -> x1 (f32)
  gemm_wmma_kernel<2><<<dim3(Dc / 128, M / 128), 256, 0, stream>>>(
      ctx, woT, bo, x1, x, 1.0f, M, Dc, Dc);

  // LN2 -> xn2 (bf16)
  layernorm_kernel<<<M, 256, 0, stream>>>(x1, g2, be2, xn2);

  // FFN up (exact GELU) -> hb (bf16)
  gemm_wmma_kernel<1><<<dim3(FFc / 128, M / 128), 256, 0, stream>>>(
      xn2, w1T, b1, hb, nullptr, 1.0f, M, FFc, Dc);

  // FFN down + residual -> d_out (f32)
  gemm_wmma_kernel<2><<<dim3(Dc / 128, M / 128), 256, 0, stream>>>(
      hb, w2T, b2, out, x1, 1.0f, M, Dc, FFc);
}
